// MultiHeadSelfAttention_27960237097586
// MI455X (gfx1250) — matrix operand order repaired and checked by reading, not compiled, not run on hardware
//
#include <hip/hip_runtime.h>

// ---------------- problem constants ----------------
#define S_LEN 2048
#define EMB   1024
#define NH    16
#define HDIM  64
#define BATCH 2
#define M_ROWS (BATCH * S_LEN)   // 4096
#define N_QKV  (3 * EMB)         // 3072

#define BLK_M  64
#define BLK_N  256
#define KSTEP  32

typedef __bf16 bf16_t;
typedef __bf16 v16bf __attribute__((ext_vector_type(16)));
typedef __bf16 v8bf  __attribute__((ext_vector_type(8)));
typedef float  v8f   __attribute__((ext_vector_type(8)));
typedef unsigned u32x4 __attribute__((ext_vector_type(4)));
typedef int      i32x8 __attribute__((ext_vector_type(8)));
typedef int      i32x4 __attribute__((ext_vector_type(4)));

// ---------------- helpers ----------------
__device__ __forceinline__ bf16_t f2b(float f) {
    union { float f; unsigned u; } x; x.f = f;
    unsigned r = x.u + 0x7FFFu + ((x.u >> 16) & 1u);   // RNE
    union { unsigned short s; bf16_t b; } y; y.s = (unsigned short)(r >> 16);
    return y.b;
}

// A operand: 16x32 (MxK). lane<16: M=lane, K={0..7,16..23}; lane>=16: M=lane-16, K={8..15,24..31}
__device__ __forceinline__ v16bf load_frag_a(const bf16_t* p, int ld) {
    int lane = threadIdx.x & 31;
    const bf16_t* q = p + (size_t)(lane & 15) * ld + ((lane >> 4) << 3);
    v8bf lo = *(const v8bf*)q;          // K = kofs .. kofs+7
    v8bf hi = *(const v8bf*)(q + 16);   // K = kofs+16 .. kofs+23
    v16bf r;
#pragma unroll
    for (int i = 0; i < 8; ++i) { r[i] = lo[i]; r[i + 8] = hi[i]; }
    return r;
}

// B operand: 32x16 (KxN) from source laid out [N][K] row-major (B = src^T).
// lane<16: N=lane, K={0..7,16..23} ; lane>=16: N=lane-16, K={8..15,24..31}.
__device__ __forceinline__ v16bf load_frag_b(const bf16_t* p, int ld) {
    int lane = threadIdx.x & 31;
    const bf16_t* q = p + (size_t)(lane & 15) * ld + ((lane >> 4) << 3);
    v8bf lo = *(const v8bf*)q;
    v8bf hi = *(const v8bf*)(q + 16);
    v16bf r;
#pragma unroll
    for (int i = 0; i < 8; ++i) { r[i] = lo[i]; r[i + 8] = hi[i]; }
    return r;
}

__device__ __forceinline__ v8f wmma_bf16(v16bf a, v16bf b, v8f c) {
    return __builtin_amdgcn_wmma_f32_16x16x32_bf16(false, a, false, b, (short)0, c,
                                                   false, false);
}

// ---------------- Tensor Data Mover: 2-D tile global -> LDS ----------------
// tile_d0 elements (bf16) along innermost dim, tile_d1 rows, row stride in elements.
// Tensor dims set equal to tile dims (tile is exactly in-bounds). data_size = 2 bytes.
__device__ __forceinline__ void tdm_load_2d(void* lds_ptr, const void* gptr,
                                            unsigned tile_d0, unsigned tile_d1,
                                            unsigned long long stride0) {
    unsigned long long ga = (unsigned long long)(uintptr_t)gptr;
    unsigned lds = (unsigned)(uintptr_t)lds_ptr;   // low 32 bits = LDS byte offset
    u32x4 g0;
    g0[0] = 1u;                                    // count=1, user descriptor
    g0[1] = lds;                                   // lds_addr
    g0[2] = (unsigned)ga;                          // global_addr[31:0]
    g0[3] = (unsigned)((ga >> 32) & 0x01FFFFFFull) | (2u << 30); // addr[56:32] | type=2
    i32x8 g1;
    g1[0] = (int)(1u << 16);                                        // data_size=1 (2B)
    g1[1] = (int)((tile_d0 & 0xFFFFu) << 16);                       // tensor_dim0 lo
    g1[2] = (int)((tile_d0 >> 16) | ((tile_d1 & 0xFFFFu) << 16));   // d0 hi | d1 lo
    g1[3] = (int)((tile_d1 >> 16) | ((tile_d0 & 0xFFFFu) << 16));   // d1 hi | tile_dim0
    g1[4] = (int)(tile_d1 & 0xFFFFu);                               // tile_dim1 (dim2=0)
    g1[5] = (int)(unsigned)stride0;                                 // dim0 stride lo
    g1[6] = (int)((unsigned)(stride0 >> 32) & 0xFFFFu);             // dim0 stride hi
    g1[7] = 0;
    i32x4 z4 = {0, 0, 0, 0};
    i32x8 z8 = {0, 0, 0, 0, 0, 0, 0, 0};
    __builtin_amdgcn_tensor_load_to_lds(g0, g1, z4, z4, z8, 0);
}

// ---------------- f32 -> bf16 conversion ----------------
__global__ __launch_bounds__(256) void cvt_kernel(const float* __restrict__ src,
                                                  bf16_t* __restrict__ dst, int n) {
    int i = blockIdx.x * blockDim.x + threadIdx.x;
    int stride = gridDim.x * blockDim.x;
    for (; i < n; i += stride) dst[i] = f2b(src[i]);
}

// ---------------- block GEMM core: 64x256 per block, TDM double-buffered ----------------
// A: [M][K] at row m0; W: [N][K] at row n0; K == EMB. Wave (wr,wc) owns 32x64.
__device__ __forceinline__ void gemm_block_tdm(const bf16_t* A, const bf16_t* W,
                                               bf16_t (*sA)[BLK_M * KSTEP],
                                               bf16_t (*sB)[BLK_N * KSTEP],
                                               v8f acc[2][4]) {
    int wid = threadIdx.x >> 5;
    int wr = wid >> 2, wc = wid & 3;
    if (wid == 0) {
        tdm_load_2d(&sA[0][0], A, KSTEP, BLK_M, EMB);
        tdm_load_2d(&sB[0][0], W, KSTEP, BLK_N, EMB);
    }
    for (int k = 0, it = 0; k < EMB; k += KSTEP, it ^= 1) {
        if (wid == 0) __builtin_amdgcn_s_wait_tensorcnt(0);
        __syncthreads();   // buffer `it` ready; everyone done reading buffer `it^1`
        if (wid == 0 && (k + KSTEP) < EMB) {
            tdm_load_2d(&sA[it ^ 1][0], A + k + KSTEP, KSTEP, BLK_M, EMB);
            tdm_load_2d(&sB[it ^ 1][0], W + k + KSTEP, KSTEP, BLK_N, EMB);
        }
        const bf16_t* pa = &sA[it][0];
        const bf16_t* pb = &sB[it][0];
        v16bf a0 = load_frag_a(pa + (size_t)(32 * wr) * KSTEP, KSTEP);
        v16bf a1 = load_frag_a(pa + (size_t)(32 * wr + 16) * KSTEP, KSTEP);
#pragma unroll
        for (int j = 0; j < 4; ++j) {
            v16bf bj = load_frag_b(pb + (size_t)(64 * wc + 16 * j) * KSTEP, KSTEP);
            acc[0][j] = wmma_bf16(a0, bj, acc[0][j]);
            acc[1][j] = wmma_bf16(a1, bj, acc[1][j]);
        }
    }
}

// ---------------- QKV projection + scatter (V stored transposed) ----------------
__global__ __launch_bounds__(256) void qkv_gemm_kernel(const bf16_t* __restrict__ Xb,
                                                       const bf16_t* __restrict__ Wq,
                                                       const float*  __restrict__ bq,
                                                       bf16_t* __restrict__ Qb,
                                                       bf16_t* __restrict__ Kb,
                                                       bf16_t* __restrict__ Vtb) {
    __shared__ alignas(64) bf16_t sA[2][BLK_M * KSTEP];   //  8 KB
    __shared__ alignas(64) bf16_t sB[2][BLK_N * KSTEP];   // 32 KB
    int m0 = blockIdx.y * BLK_M;
    int n0 = blockIdx.x * BLK_N;

    v8f acc[2][4];
#pragma unroll
    for (int i = 0; i < 2; ++i)
#pragma unroll
        for (int j = 0; j < 4; ++j)
#pragma unroll
            for (int r = 0; r < 8; ++r) acc[i][j][r] = 0.0f;

    gemm_block_tdm(Xb + (size_t)m0 * EMB, Wq + (size_t)n0 * EMB, sA, sB, acc);

    int wid = threadIdx.x >> 5;
    int wr = wid >> 2, wc = wid & 3;
    int lane = threadIdx.x & 31, hh = lane >> 4, nn = lane & 15;
#pragma unroll
    for (int i = 0; i < 2; ++i)
#pragma unroll
        for (int j = 0; j < 4; ++j)
#pragma unroll
            for (int r = 0; r < 8; ++r) {
                int m = m0 + 32 * wr + 16 * i + r + 8 * hh;
                int n = n0 + 64 * wc + 16 * j + nn;
                float v = acc[i][j][r] + bq[n];
                int b = m >> 11, s = m & (S_LEN - 1);
                int h = n / (3 * HDIM), c = n % (3 * HDIM);
                size_t bh = (size_t)(b * NH + h);
                if (c < HDIM)
                    Qb[(bh * S_LEN + s) * HDIM + c] = f2b(v);
                else if (c < 2 * HDIM)
                    Kb[(bh * S_LEN + s) * HDIM + (c - HDIM)] = f2b(v);
                else
                    Vtb[(bh * HDIM + (c - 2 * HDIM)) * S_LEN + s] = f2b(v);
            }
}

// ---------------- flash attention: 4 waves/block share (b,h); K/V via TDM ----------------
__global__ __launch_bounds__(128) void attn_kernel(const bf16_t* __restrict__ Qb,
                                                   const bf16_t* __restrict__ Kb,
                                                   const bf16_t* __restrict__ Vtb,
                                                   bf16_t* __restrict__ Ctx) {
    __shared__ alignas(64) bf16_t sK[2][32 * HDIM];   // 2 x 4 KB, [t][d]
    __shared__ alignas(64) bf16_t sV[2][HDIM * 32];   // 2 x 4 KB, [d][t]
    __shared__ alignas(64) bf16_t pls[4 * 16 * 32];   // 4 KB, wave-private P tiles

    int wiw = threadIdx.x >> 5;
    int w   = blockIdx.x * 4 + wiw;
    int mblk = w % (S_LEN / 16);
    int h    = (w / (S_LEN / 16)) % NH;
    int b    = w / ((S_LEN / 16) * NH);
    int m0   = mblk * 16;
    size_t bh = (size_t)(b * NH + h);
    const bf16_t* Qp = Qb  + (bh * S_LEN + m0) * HDIM;
    const bf16_t* Kp = Kb  + bh * S_LEN * HDIM;
    const bf16_t* Vp = Vtb + bh * HDIM * S_LEN;
    bf16_t* lbase = pls + wiw * (16 * 32);

    v16bf qa0 = load_frag_a(Qp, HDIM);        // K-dim chunk 0..31
    v16bf qa1 = load_frag_a(Qp + 32, HDIM);   // K-dim chunk 32..63

    v8f mrow, lrow, ctx[4];
#pragma unroll
    for (int r = 0; r < 8; ++r) { mrow[r] = -3.0e38f; lrow[r] = 0.0f; }
#pragma unroll
    for (int j = 0; j < 4; ++j)
#pragma unroll
        for (int r = 0; r < 8; ++r) ctx[j][r] = 0.0f;

    int lane = threadIdx.x & 31, hh = lane >> 4, nn = lane & 15;
    const float scale = 0.125f;   // 1/sqrt(64)

    if (wiw == 0) {
        // K tile rows are contiguous (row stride == HDIM): one 4 KB linear tile.
        tdm_load_2d(&sK[0][0], Kp, 32 * HDIM, 1, 32 * HDIM);
        tdm_load_2d(&sV[0][0], Vp, 32, HDIM, S_LEN);
    }

    for (int t0 = 0, it = 0; t0 < S_LEN; t0 += 32, it ^= 1) {
        if (wiw == 0) __builtin_amdgcn_s_wait_tensorcnt(0);
        __syncthreads();
        if (wiw == 0 && (t0 + 32) < S_LEN) {
            tdm_load_2d(&sK[it ^ 1][0], Kp + (size_t)(t0 + 32) * HDIM, 32 * HDIM, 1,
                        32 * HDIM);
            tdm_load_2d(&sV[it ^ 1][0], Vp + (t0 + 32), 32, HDIM, S_LEN);
        }
        const bf16_t* kt = &sK[it][0];
        const bf16_t* vt = &sV[it][0];

        // ---- scores: 16 queries x 32 keys, two 16x16 tiles ----
        v8f s0, s1;
#pragma unroll
        for (int r = 0; r < 8; ++r) { s0[r] = 0.0f; s1[r] = 0.0f; }
        s0 = wmma_bf16(qa0, load_frag_b(kt, HDIM), s0);
        s0 = wmma_bf16(qa1, load_frag_b(kt + 32, HDIM), s0);
        s1 = wmma_bf16(qa0, load_frag_b(kt + 16 * HDIM, HDIM), s1);
        s1 = wmma_bf16(qa1, load_frag_b(kt + 16 * HDIM + 32, HDIM), s1);

        // ---- online softmax (row stats live in the same fragment slots) ----
        v8f mx;
#pragma unroll
        for (int r = 0; r < 8; ++r) {
            s0[r] *= scale; s1[r] *= scale;
            mx[r] = fmaxf(s0[r], s1[r]);
        }
#pragma unroll
        for (int msk = 1; msk < 16; msk <<= 1)
#pragma unroll
            for (int r = 0; r < 8; ++r)
                mx[r] = fmaxf(mx[r], __shfl_xor(mx[r], msk, 32));

        v8f mnew, corr, ps;
#pragma unroll
        for (int r = 0; r < 8; ++r) {
            mnew[r] = fmaxf(mrow[r], mx[r]);
            corr[r] = __expf(mrow[r] - mnew[r]);
            s0[r]   = __expf(s0[r] - mnew[r]);
            s1[r]   = __expf(s1[r] - mnew[r]);
            ps[r]   = s0[r] + s1[r];
        }
#pragma unroll
        for (int msk = 1; msk < 16; msk <<= 1)
#pragma unroll
            for (int r = 0; r < 8; ++r)
                ps[r] += __shfl_xor(ps[r], msk, 32);
#pragma unroll
        for (int r = 0; r < 8; ++r) {
            lrow[r] = lrow[r] * corr[r] + ps[r];
            mrow[r] = mnew[r];
        }
#pragma unroll
        for (int j = 0; j < 4; ++j)
#pragma unroll
            for (int r = 0; r < 8; ++r) ctx[j][r] *= corr[r];

        // ---- transpose P through wave-private LDS into A-fragment layout ----
#pragma unroll
        for (int r = 0; r < 8; ++r) {
            int row = r + 8 * hh;
            lbase[row * 32 + nn]      = f2b(s0[r]);
            lbase[row * 32 + 16 + nn] = f2b(s1[r]);
        }
        asm volatile("s_wait_dscnt 0" ::: "memory");
        v16bf pa = load_frag_a(lbase, 32);

        // ---- ctx += P @ V  (V^T tile rows contiguous along t in LDS) ----
#pragma unroll
        for (int j = 0; j < 4; ++j) {
            v16bf vb = load_frag_b(vt + (size_t)(16 * j) * 32, 32);
            ctx[j] = wmma_bf16(pa, vb, ctx[j]);
        }
    }

    // ---- normalize and store ctx (bf16, [b][s][h*HD+d]) ----
    v8f inv;
#pragma unroll
    for (int r = 0; r < 8; ++r) inv[r] = 1.0f / lrow[r];
#pragma unroll
    for (int j = 0; j < 4; ++j)
#pragma unroll
        for (int r = 0; r < 8; ++r) {
            int m = m0 + r + 8 * hh;
            Ctx[((size_t)b * S_LEN + m) * EMB + h * HDIM + 16 * j + nn] =
                f2b(ctx[j][r] * inv[r]);
        }
}

// ---------------- output projection -> fp32 d_out ----------------
__global__ __launch_bounds__(256) void out_gemm_kernel(const bf16_t* __restrict__ Ctx,
                                                       const bf16_t* __restrict__ Wo,
                                                       const float*  __restrict__ bo,
                                                       float* __restrict__ out) {
    __shared__ alignas(64) bf16_t sA[2][BLK_M * KSTEP];
    __shared__ alignas(64) bf16_t sB[2][BLK_N * KSTEP];
    int m0 = blockIdx.y * BLK_M;
    int n0 = blockIdx.x * BLK_N;

    v8f acc[2][4];
#pragma unroll
    for (int i = 0; i < 2; ++i)
#pragma unroll
        for (int j = 0; j < 4; ++j)
#pragma unroll
            for (int r = 0; r < 8; ++r) acc[i][j][r] = 0.0f;

    gemm_block_tdm(Ctx + (size_t)m0 * EMB, Wo + (size_t)n0 * EMB, sA, sB, acc);

    int wid = threadIdx.x >> 5;
    int wr = wid >> 2, wc = wid & 3;
    int lane = threadIdx.x & 31, hh = lane >> 4, nn = lane & 15;
#pragma unroll
    for (int i = 0; i < 2; ++i)
#pragma unroll
        for (int j = 0; j < 4; ++j)
#pragma unroll
            for (int r = 0; r < 8; ++r) {
                int m = m0 + 32 * wr + 16 * i + r + 8 * hh;
                int n = n0 + 64 * wc + 16 * j + nn;
                out[(size_t)m * EMB + n] = acc[i][j][r] + bo[n];
            }
}

// ---------------- host-side launch ----------------
extern "C" void kernel_launch(void* const* d_in, const int* in_sizes, int n_in,
                              void* d_out, int out_size, void* d_ws, size_t ws_size,
                              hipStream_t stream) {
    const float* x     = (const float*)d_in[0];
    const float* Wqkv  = (const float*)d_in[1];
    const float* bqkv  = (const float*)d_in[2];
    const float* Wout  = (const float*)d_in[3];
    const float* bout  = (const float*)d_in[4];
    float*       out   = (float*)d_out;

    const int XN  = M_ROWS * EMB;
    const int WQN = N_QKV * EMB;
    const int WON = EMB * EMB;
    const int QN  = BATCH * NH * S_LEN * HDIM;
    const int CN  = M_ROWS * EMB;

    char* ws = (char*)d_ws;
    size_t off = 0;
    auto take = [&](size_t elems) -> bf16_t* {
        bf16_t* p = (bf16_t*)(ws + off);
        off = (off + elems * sizeof(bf16_t) + 255) & ~(size_t)255;
        return p;
    };
    bf16_t* Xb    = take(XN);
    bf16_t* Wqkvb = take(WQN);
    bf16_t* Woutb = take(WON);
    bf16_t* Qb    = take(QN);
    bf16_t* Kb    = take(QN);
    bf16_t* Vtb   = take(QN);
    bf16_t* Ctxb  = take(CN);
    (void)ws_size; (void)in_sizes; (void)n_in; (void)out_size;

    // 1) convert to bf16
    cvt_kernel<<<(XN  + 255) / 256, 256, 0, stream>>>(x,    Xb,    XN);
    cvt_kernel<<<(WQN + 255) / 256, 256, 0, stream>>>(Wqkv, Wqkvb, WQN);
    cvt_kernel<<<(WON + 255) / 256, 256, 0, stream>>>(Wout, Woutb, WON);

    // 2) QKV projection: blocks of 64x256 over (4096 x 3072)
    qkv_gemm_kernel<<<dim3(N_QKV / BLK_N, M_ROWS / BLK_M), 256, 0, stream>>>(
        Xb, Wqkvb, bqkv, Qb, Kb, Vtb);

    // 3) attention: waves = B*H*(S/16) = 4096 -> 1024 blocks of 4 waves
    attn_kernel<<<1024, 128, 0, stream>>>(Qb, Kb, Vtb, Ctxb);

    // 4) output projection: blocks of 64x256 over (4096 x 1024)
    out_gemm_kernel<<<dim3(EMB / BLK_N, M_ROWS / BLK_M), 256, 0, stream>>>(
        Ctxb, Woutb, bout, out);
}
